// AutoCorrelation_45423574122699
// MI455X (gfx1250) — matrix operand order repaired and checked by reading, not compiled, not run on hardware
//
#include <hip/hip_runtime.h>
#include <math.h>

// Problem constants (B, L, H, E) = (8, 4096, 8, 64); channels C = H*E = 512.
#define B_    8
#define L_    4096
#define C_    512
#define BLK   128                 // Q/K rows per block side
#define TBLK  (L_ / BLK)          // 32 blocks per side
#define NWG   (B_ * TBLK * TBLK)  // 8192 correlation blocks
#define CC    64                  // channels staged into LDS per iteration
#define STR   72                  // LDS row stride in bf16 elems (64 + 8 pad -> no bank conflicts)
#define TOPK  8                   // int(log(4096)) == 8
#define NBIN  255                 // distinct diagonal offsets per 128x128 block
#define BINP  272                 // padded bin row

typedef __attribute__((ext_vector_type(16))) __bf16       v16bf;
typedef __attribute__((ext_vector_type(8)))  float        v8f;
typedef __attribute__((ext_vector_type(4)))  unsigned int u32x4;
typedef __attribute__((ext_vector_type(2)))  unsigned int u32x2;
typedef __attribute__((ext_vector_type(4)))  float        f32x4;

struct U32x8 { u32x4 lo, hi; };

__device__ __forceinline__ unsigned int bf16_rne(float x) {
  unsigned int u = __float_as_uint(x);
  return (u + 0x7fffu + ((u >> 16) & 1u)) >> 16;   // round-to-nearest-even
}
__device__ __forceinline__ unsigned int pack2(float a, float b) {
  return bf16_rne(a) | (bf16_rne(b) << 16);
}
__device__ __forceinline__ v16bf mkfrag(u32x4 lo, u32x4 hi) {
  U32x8 t{lo, hi};
  return __builtin_bit_cast(v16bf, t);
}

// ---------------------------------------------------------------------------
// Kernel 1: blocked Gram (Q.K^T) via v_wmma_f32_16x16x32_bf16, reduced along
// circular diagonals into 255 per-block partial sums. Fully deterministic:
// binning uses per-wave, per-half LDS rows so every concurrent store has a
// unique address (plain RMW, no atomics).
// ---------------------------------------------------------------------------
__global__ __launch_bounds__(256, 2)
void corr_wmma_kernel(const float* __restrict__ q, const float* __restrict__ k,
                      float* __restrict__ part)
{
  __shared__ __align__(16) unsigned short Qs[BLK * STR];   // 18 KB
  __shared__ __align__(16) unsigned short Ks[BLK * STR];   // 18 KB
  __shared__ float bins[8][2][BINP];                       // 17 KB

  const int tid  = threadIdx.x;
  const int wv   = tid >> 5;    // wave id == tile-row tT (0..7)
  const int ln   = tid & 31;
  const int half = ln >> 4;
  const int n    = ln & 15;     // N column (B/C/D lane layout)

  const int wg = blockIdx.x;
  const int b  = wg >> 10;
  const int Tb = (wg >> 5) & 31;
  const int Sb = wg & 31;

  const float* qb = q + (size_t)(b * L_ + Tb * BLK) * C_;
  const float* kb = k + (size_t)(b * L_ + Sb * BLK) * C_;

  for (int i = tid; i < 8 * 2 * BINP; i += 256) (&bins[0][0][0])[i] = 0.0f;

  v8f acc[8];
#pragma unroll
  for (int s = 0; s < 8; ++s)
#pragma unroll
    for (int j = 0; j < 8; ++j) acc[s][j] = 0.0f;

  for (int cc = 0; cc < C_; cc += CC) {
    __syncthreads();
    // Stage 128 rows x 64 channels of Q and K, f32 -> bf16 (RNE), into LDS.
#pragma unroll
    for (int g = 0; g < 8; ++g) {
      int idx = tid + (g << 8);          // 0..2047
      int row = idx >> 4;                // 0..127
      int c4  = (idx & 15) << 2;         // 0,4,...,60
      f32x4 vq = *(const f32x4*)(qb + (size_t)row * C_ + cc + c4);
      f32x4 vk = *(const f32x4*)(kb + (size_t)row * C_ + cc + c4);
      u32x2 pq; pq.x = pack2(vq.x, vq.y); pq.y = pack2(vq.z, vq.w);
      u32x2 pk; pk.x = pack2(vk.x, vk.y); pk.y = pack2(vk.z, vk.w);
      *(u32x2*)&Qs[row * STR + c4] = pq;
      *(u32x2*)&Ks[row * STR + c4] = pk;
    }
    if (cc + CC < C_) {                  // global_prefetch_b8 for next stage
      __builtin_prefetch(qb + (size_t)(tid >> 1) * C_ + cc + CC, 0, 1);
      __builtin_prefetch(kb + (size_t)(tid >> 1) * C_ + cc + CC, 0, 1);
    }
    __syncthreads();

#pragma unroll
    for (int kc = 0; kc < CC / 32; ++kc) {
      // A fragment (ISA 16-bit A 16x32): lane=M row, per-half 8-chan blocks.
      const int arow = wv * 16 + n;
      const int aoff = arow * STR + kc * 32 + half * 8;
      v16bf av = mkfrag(*(const u32x4*)&Qs[aoff],
                        *(const u32x4*)&Qs[aoff + 16]);
#pragma unroll
      for (int sS = 0; sS < 8; ++sS) {
        // B fragment (32x16, K x N): lane=N, per-half 8-chan blocks (same K order as A).
        const int brow = sS * 16 + n;
        const int boff = brow * STR + kc * 32 + half * 8;
        v16bf bv = mkfrag(*(const u32x4*)&Ks[boff],
                          *(const u32x4*)&Ks[boff + 16]);
        acc[sS] = __builtin_amdgcn_wmma_f32_16x16x32_bf16(
            false, av, false, bv, (short)0, acc[sS], false, false);
      }
    }
  }

  // Diagonal binning. C/D layout: entry j of lane -> (m = j + 8*half, n).
  // bin = (tT - sS)*16 + (m - n) + 127  in [0, 254].
#pragma unroll
  for (int sS = 0; sS < 8; ++sS) {
    const int base = (wv - sS) * 16 + 127 - n + (half ? 8 : 0);
#pragma unroll
    for (int j = 0; j < 8; ++j) {
      bins[wv][half][base + j] += acc[sS][j];   // unique addr per lane: no race
    }
  }
  __syncthreads();

  float s = 0.0f;
  if (tid < NBIN) {
#pragma unroll
    for (int w2 = 0; w2 < 8; ++w2) s += bins[w2][0][tid] + bins[w2][1][tid];
  }
  part[(size_t)wg * 256 + tid] = (tid < NBIN) ? s : 0.0f;
}

// ---------------------------------------------------------------------------
// Kernel 2: deterministic reduction of per-block diagonal partials into
// mean_corr[B, L] (includes the 1/512 mean over H*E channels).
// ---------------------------------------------------------------------------
__global__ __launch_bounds__(256)
void mc_reduce_kernel(const float* __restrict__ part, float* __restrict__ mc)
{
  int gid = blockIdx.x * 256 + threadIdx.x;
  if (gid >= B_ * L_) return;
  int b   = gid >> 12;
  int tau = gid & (L_ - 1);
  const float* pb = part + (size_t)b * TBLK * TBLK * 256;
  float s = 0.0f;
  for (int d = 0; d < TBLK; ++d) {
    int i = (tau - 128 * d + 127 + L_) & (L_ - 1);
    if (i < NBIN) {
      for (int Tb = 0; Tb < TBLK; ++Tb) {
        int Sb = (Tb - d) & (TBLK - 1);
        s += pb[((size_t)Tb * TBLK + Sb) * 256 + i];
      }
    }
  }
  mc[gid] = s * (1.0f / 512.0f);
}

// ---------------------------------------------------------------------------
// Kernel 3: per-batch top-8 (8 argmax tree reductions) + softmax.
// ---------------------------------------------------------------------------
__global__ __launch_bounds__(256)
void topk_softmax_kernel(const float* __restrict__ mc,
                         int* __restrict__ idxo, float* __restrict__ wo)
{
  __shared__ float vals[L_];
  __shared__ float rv[256];
  __shared__ int   ri[256];
  __shared__ float tv[TOPK];
  __shared__ int   ti[TOPK];
  const int b = blockIdx.x, tid = threadIdx.x;
  for (int i = tid; i < L_; i += 256) vals[i] = mc[b * L_ + i];
  __syncthreads();
  for (int kk = 0; kk < TOPK; ++kk) {
    float bv = -3.402823466e38f; int bi = L_;
    for (int i = tid; i < L_; i += 256) {
      float x = vals[i];
      if (x > bv || (x == bv && i < bi)) { bv = x; bi = i; }
    }
    rv[tid] = bv; ri[tid] = bi;
    __syncthreads();
    for (int stp = 128; stp > 0; stp >>= 1) {
      if (tid < stp) {
        float ov = rv[tid + stp]; int oi = ri[tid + stp];
        if (ov > rv[tid] || (ov == rv[tid] && oi < ri[tid])) {
          rv[tid] = ov; ri[tid] = oi;
        }
      }
      __syncthreads();
    }
    if (tid == 0) { tv[kk] = rv[0]; ti[kk] = ri[0]; vals[ri[0]] = -3.402823466e38f; }
    __syncthreads();
  }
  if (tid == 0) {
    float mx = tv[0];
    for (int i = 1; i < TOPK; ++i) mx = fmaxf(mx, tv[i]);
    float e[TOPK], sum = 0.0f;
    for (int i = 0; i < TOPK; ++i) { e[i] = expf(tv[i] - mx); sum += e[i]; }
    float inv = 1.0f / sum;
    for (int i = 0; i < TOPK; ++i) {
      wo[b * TOPK + i]   = e[i] * inv;
      idxo[b * TOPK + i] = ti[i];
    }
  }
}

// ---------------------------------------------------------------------------
// Kernel 4: out[b,t,:] = sum_i w[b,i] * V[b, (t + delay_i) mod L, :]
// one block per (b, t); 128 threads x float4 = 512 channels.
// ---------------------------------------------------------------------------
__global__ __launch_bounds__(128)
void delay_agg_kernel(const float* __restrict__ v, const int* __restrict__ idxo,
                      const float* __restrict__ wo, float* __restrict__ out)
{
  __shared__ float wsm[TOPK];
  __shared__ int   dsm[TOPK];
  const int blk = blockIdx.x;
  const int b = blk >> 12;
  const int t = blk & (L_ - 1);
  if (threadIdx.x < TOPK) {
    wsm[threadIdx.x] = wo[b * TOPK + threadIdx.x];
    dsm[threadIdx.x] = idxo[b * TOPK + threadIdx.x];
  }
  __syncthreads();
  const float* vb = v + (size_t)b * L_ * C_;
  const int c = threadIdx.x << 2;
  f32x4 a; a.x = a.y = a.z = a.w = 0.0f;
#pragma unroll
  for (int i = 0; i < TOPK; ++i) {
    int src = (t + dsm[i]) & (L_ - 1);
    f32x4 x = *(const f32x4*)(vb + (size_t)src * C_ + c);
    float wi = wsm[i];
    a.x += wi * x.x; a.y += wi * x.y; a.z += wi * x.z; a.w += wi * x.w;
  }
  *(f32x4*)(out + ((size_t)b * L_ + t) * C_ + c) = a;
}

// ---------------------------------------------------------------------------
extern "C" void kernel_launch(void* const* d_in, const int* in_sizes, int n_in,
                              void* d_out, int out_size, void* d_ws, size_t ws_size,
                              hipStream_t stream) {
  (void)in_sizes; (void)n_in; (void)out_size; (void)ws_size;
  const float* q = (const float*)d_in[0];
  const float* k = (const float*)d_in[1];
  const float* v = (const float*)d_in[2];
  float* out = (float*)d_out;

  char* ws = (char*)d_ws;
  float* part = (float*)ws;                                   // NWG*256 f32 (8 MiB)
  size_t off  = (size_t)NWG * 256 * sizeof(float);
  float* mc   = (float*)(ws + off);  off += (size_t)B_ * L_ * sizeof(float);
  int*   tix  = (int*)  (ws + off);  off += (size_t)B_ * TOPK * sizeof(int);
  float* tw   = (float*)(ws + off);

  corr_wmma_kernel<<<NWG, 256, 0, stream>>>(q, k, part);
  mc_reduce_kernel<<<(B_ * L_ + 255) / 256, 256, 0, stream>>>(part, mc);
  topk_softmax_kernel<<<B_, 256, 0, stream>>>(mc, tix, tw);
  delay_agg_kernel<<<B_ * L_, 128, 0, stream>>>(v, tix, tw, out);
}
